// LiquidNeuralNetwork_6768868458799
// MI455X (gfx1250) — compile-verified, matrix operand order checked by reading, not run on hardware
//
#include <hip/hip_runtime.h>
#include <hip/hip_bf16.h>

// LiquidNeuralNetwork scan for gfx1250 (MI455X), wave32 + WMMA.
// B=256, S=4096, I=32, H=64, O=1.

typedef __attribute__((ext_vector_type(16))) _Float16 v16h;
typedef __attribute__((ext_vector_type(8)))  float    v8f;

#define LNN_B 256
#define LNN_S 4096
#define LNN_I 32
#define LNN_H 64

static __device__ __forceinline__ v8f wmma32f16(v16h a, v16h b, v8f c) {
  // (neg_a, A, neg_b, B, c_mod, C, reuse_a, reuse_b)
  return __builtin_amdgcn_wmma_f32_16x16x32_f16(false, a, false, b, (short)0, c,
                                                false, false);
}

static __device__ __forceinline__ float fast_tanh(float x) {
  // tanh(x) = 1 - 2/(exp(2x)+1); clamp to avoid inf/inf. Uses HW v_exp_f32.
  float xx = fminf(fmaxf(x, -9.0f), 9.0f);
  float e  = __expf(2.0f * xx);
  return 1.0f - 2.0f / (e + 1.0f);
}

// 16-bit A/B fragment K mapping (CDNA5 ISA 7.12.2):
// element e in lane: e<8 -> K = e + 8*half ; e>=8 -> K = 16 + (e-8) + 8*half
static __device__ __forceinline__ int kmap(int e, int half) {
  return (e < 8) ? (e + 8 * half) : (16 + (e - 8) + 8 * half);
}

__global__ __launch_bounds__(32, 1) void lnn_scan_wmma(
    const float* __restrict__ x,     // [B,S,I]
    const float* __restrict__ W_in,  // [H,I]
    const float* __restrict__ b_in,  // [H]
    const float* __restrict__ W_hh,  // [H,H]
    const float* __restrict__ W_ih,  // [H,H]
    const float* __restrict__ bias,  // [H]
    const float* __restrict__ tau,   // [H]
    const float* __restrict__ W_out, // [O,H] = [1,64]
    const float* __restrict__ b_out, // [1]
    float* __restrict__ out)         // [B,S,1]
{
  __shared__ float sWin[LNN_H * LNN_I];
  __shared__ float sbin[LNN_H];

  const int lane  = threadIdx.x & 31;
  const int col   = lane & 15;   // batch column within tile (N) / A-row (M)
  const int half  = lane >> 4;   // lane half selects K sub-range
  const int bbase = blockIdx.x * 16;

  // Stage W_in / b_in in LDS for the W_comb = W_ih @ W_in fold.
  for (int i = lane; i < LNN_H * LNN_I; i += 32) sWin[i] = W_in[i];
  for (int i = lane; i < LNN_H; i += 32)         sbin[i] = b_in[i];
  __syncthreads();

  // ---- Constant A fragments (packed once, live in VGPRs) ----
  // a_ie[mt]    : W_comb[m][k]      (16x32, one WMMA K-step),  m = col + 16*mt
  // a_hh[kt][mt]: W_hh[m][k+32*kt]  (16x64 split into 2 K-steps)
  v16h a_ie[4];
  v16h a_hh[2][4];
  #pragma unroll
  for (int mt = 0; mt < 4; ++mt) {
    const int m = col + 16 * mt;
    #pragma unroll
    for (int e = 0; e < 16; ++e) {
      const int k = kmap(e, half);
      float acc = 0.0f;
      for (int hh = 0; hh < LNN_H; ++hh)
        acc += W_ih[m * LNN_H + hh] * sWin[hh * LNN_I + k];
      a_ie[mt][e]    = (_Float16)acc;
      a_hh[0][mt][e] = (_Float16)W_hh[m * LNN_H + k];
      a_hh[1][mt][e] = (_Float16)W_hh[m * LNN_H + k + 32];
    }
  }

  // ---- Per-lane update constants, indexed by (mt, v): hidden m = v + 8*half + 16*mt
  float itau_r[32], btau_r[32], wout_r[32];
  #pragma unroll
  for (int mt = 0; mt < 4; ++mt) {
    #pragma unroll
    for (int v = 0; v < 8; ++v) {
      const int m = v + 8 * half + 16 * mt;
      float it = 1.0f / tau[m];
      float be = bias[m];                     // effective bias: bias + W_ih @ b_in
      for (int hh = 0; hh < LNN_H; ++hh) be += W_ih[m * LNN_H + hh] * sbin[hh];
      itau_r[mt * 8 + v] = it;
      btau_r[mt * 8 + v] = be * it;
      wout_r[mt * 8 + v] = W_out[m];
    }
  }

  // ---- Register-resident scan state (transposed: lane = batch col, vgpr = hidden)
  v8f hst[4], th[4];
  #pragma unroll
  for (int mt = 0; mt < 4; ++mt) {
    #pragma unroll
    for (int v = 0; v < 8; ++v) { hst[mt][v] = 0.0f; th[mt][v] = 0.0f; }
  }

  const float* __restrict__ xrow = x + (size_t)(bbase + col) * LNN_S * LNN_I;
  float* __restrict__ orow = out + (size_t)(bbase + col) * LNN_S;
  const float bo = b_out[0];

  // Software pipeline: preload x_t fragment for t=0 (coalesced b128 loads).
  const float* xp0 = xrow + 8 * half;
  float4 nx0 = *(const float4*)(xp0);
  float4 nx1 = *(const float4*)(xp0 + 4);
  float4 nx2 = *(const float4*)(xp0 + 16);
  float4 nx3 = *(const float4*)(xp0 + 20);

  for (int t = 0; t < LNN_S; ++t) {
    float4 x0 = nx0, x1 = nx1, x2 = nx2, x3 = nx3;
    if (t + 1 < LNN_S) {
      const float* xp = xrow + (size_t)(t + 1) * LNN_I + 8 * half;
      nx0 = *(const float4*)(xp);
      nx1 = *(const float4*)(xp + 4);
      nx2 = *(const float4*)(xp + 16);
      nx3 = *(const float4*)(xp + 20);
    }
    {
      const int tp = (t + 8 < LNN_S) ? (t + 8) : (LNN_S - 1);
      __builtin_prefetch(xrow + (size_t)tp * LNN_I + 8 * half, 0, 0);
    }

    // B operand for input path: x_t^T (K=32, N=16 batch cols)
    v16h bx;
    bx[0]  = (_Float16)x0.x; bx[1]  = (_Float16)x0.y;
    bx[2]  = (_Float16)x0.z; bx[3]  = (_Float16)x0.w;
    bx[4]  = (_Float16)x1.x; bx[5]  = (_Float16)x1.y;
    bx[6]  = (_Float16)x1.z; bx[7]  = (_Float16)x1.w;
    bx[8]  = (_Float16)x2.x; bx[9]  = (_Float16)x2.y;
    bx[10] = (_Float16)x2.z; bx[11] = (_Float16)x2.w;
    bx[12] = (_Float16)x3.x; bx[13] = (_Float16)x3.y;
    bx[14] = (_Float16)x3.z; bx[15] = (_Float16)x3.w;

    // B operands for recurrence: tanh(h)^T, repacked D->B *within lane*
    // (batch-as-N makes the lane's column identical between D and B layouts).
    v16h bh0, bh1;
    #pragma unroll
    for (int j = 0; j < 8; ++j) {
      bh0[j]     = (_Float16)th[0][j];
      bh0[j + 8] = (_Float16)th[1][j];
      bh1[j]     = (_Float16)th[2][j];
      bh1[j + 8] = (_Float16)th[3][j];
    }

    float op = 0.0f;
    #pragma unroll
    for (int mt = 0; mt < 4; ++mt) {
      v8f d = {};
      d = wmma32f16(a_ie[mt],    bx,  d);   // ie_t^T  = W_comb @ x_t^T
      d = wmma32f16(a_hh[0][mt], bh0, d);   // + W_hh[:, 0:32]  @ tanh(h)^T
      d = wmma32f16(a_hh[1][mt], bh1, d);   // + W_hh[:,32:64]  @ tanh(h)^T
      #pragma unroll
      for (int v = 0; v < 8; ++v) {
        float h  = hst[mt][v];
        // h += (d + bias_eff - h)/tau   (dt = 1)
        float hn = h + (d[v] - h) * itau_r[mt * 8 + v] + btau_r[mt * 8 + v];
        hst[mt][v] = hn;
        float tv = fast_tanh(hn);
        th[mt][v] = tv;                                   // feeds step t+1
        op = fmaf(tv, wout_r[mt * 8 + v], op);            // output proj (O=1)
      }
    }
    // Each half holds a disjoint set of hidden rows for batch col `col`.
    op += __shfl_xor(op, 16, 32);
    if (half == 0) orow[t] = op + bo;
  }
}

extern "C" void kernel_launch(void* const* d_in, const int* in_sizes, int n_in,
                              void* d_out, int out_size, void* d_ws, size_t ws_size,
                              hipStream_t stream) {
  (void)in_sizes; (void)n_in; (void)out_size; (void)d_ws; (void)ws_size;
  const float* x     = (const float*)d_in[0];
  const float* W_in  = (const float*)d_in[1];
  const float* b_in  = (const float*)d_in[2];
  const float* W_hh  = (const float*)d_in[3];
  const float* W_ih  = (const float*)d_in[4];
  const float* bias  = (const float*)d_in[5];
  const float* tau   = (const float*)d_in[6];
  const float* W_out = (const float*)d_in[7];
  const float* b_out = (const float*)d_in[8];
  float* out = (float*)d_out;

  dim3 grid(LNN_B / 16), block(32);
  hipLaunchKernelGGL(lnn_scan_wmma, grid, block, 0, stream,
                     x, W_in, b_in, W_hh, W_ih, bias, tau, W_out, b_out, out);
}
